// OsuModel_33294586479136
// MI455X (gfx1250) — compile-verified, matrix operand order checked by reading, not run on hardware
//
#include <hip/hip_runtime.h>
#include <math.h>

typedef __bf16 bf16;
typedef __attribute__((ext_vector_type(16))) __bf16 v16bf;
typedef __attribute__((ext_vector_type(8)))  float  v8f;
typedef __attribute__((ext_vector_type(4)))  unsigned int u32x4;
typedef __attribute__((ext_vector_type(4)))  int i32x4;
typedef __attribute__((ext_vector_type(8)))  int i32x8;

#define DEV __device__ __forceinline__

constexpr int BSZ = 32, TLEN = 1024, SLEN = 128, DM = 512, INDIM = 80;
constexpr int NHEAD = 2, DATT = 512, CCH = 32, CKER = 31, PREDIM = 256;
constexpr int KMAX = 1024; // max GEMM K in this model
constexpr int KPAD = 96;   // encoder projection K (80) padded to a multiple of 32

DEV float bf2f(bf16 x) { return (float)x; }
DEV bf16  f2bf(float x) { return (bf16)x; }
DEV float sigf(float x) { return 1.f / (1.f + __expf(-x)); }

union FragU { u32x4 u[2]; v16bf v; };

DEV v16bf frag_zero() {
  v16bf f;
#pragma unroll
  for (int j = 0; j < 16; ++j) f[j] = f2bf(0.f);
  return f;
}

// Vectorized 16x32 bf16 fragment load from row-major memory (K%32==0 everywhere).
DEV v16bf load_frag_g(const bf16* __restrict__ base, int row0, int ld, int lane) {
  int n = lane & 15, kq = (lane >> 4) << 3;
  const bf16* r = base + (long)(row0 + n) * ld;
  FragU ua;
  ua.u[0] = *(const u32x4*)(r + kq);
  ua.u[1] = *(const u32x4*)(r + 16 + kq);
  return ua.v;
}

DEV v8f wmma_bf16(v16bf a, v16bf b, v8f c) {
  return __builtin_amdgcn_wmma_f32_16x16x32_bf16(false, a, false, b, (short)0, c, false, false);
}

// ---------------------------------------------------------------------------
// Generic GEMM: C[M,N](f32) = act( [acc?C:0] + A[M,K]bf16 * W[N,K]^T bf16 + bias )
// The whole 16xK A-slab for this block is DMA'd into LDS once by the Tensor
// Data Mover (tensor_load_to_lds + s_wait_tensorcnt); the K-loop then runs with
// no barriers or branches: ds_load_b128 (A) + global_load_b128 (W) + v_wmma.
// flags: bit0 = accumulate into C; bits[4:2] = act (0 none,1 relu,2 leaky0.1,3 sigmoid)
// Requires: M%16==0, N%16==0, K%32==0, K<=KMAX.
// ---------------------------------------------------------------------------
__global__ void __launch_bounds__(128)
gemm_bf16_wmma(const bf16* __restrict__ A, int lda,
               const bf16* __restrict__ W, int ldw,
               const float* __restrict__ bias,
               float* __restrict__ C, int ldc,
               bf16* __restrict__ Cb, int ldcb,
               int M, int N, int K, int flags) {
  __shared__ bf16 Aslab[16 * KMAX]; // 32 KB, packed [16][K]
  int tid = threadIdx.x;
  int lane = tid & 31, wave = tid >> 5;
  int mt = blockIdx.y;
  int nt = blockIdx.x * 4 + wave;
  int ntiles = N >> 4;
  int m = lane & 15, kq = (lane >> 4) << 3;

#if __has_builtin(__builtin_amdgcn_tensor_load_to_lds)
  if (wave == 0) {
    unsigned long long ga = (unsigned long long)(const void*)(A + (long)mt * 16 * lda);
    unsigned lds = (unsigned)(size_t)(&Aslab[0]);
    u32x4 g0;
    g0[0] = 1u;                                        // count=1 (valid user D#)
    g0[1] = lds;                                       // lds_addr
    g0[2] = (unsigned)(ga & 0xFFFFFFFFu);              // global_addr[31:0]
    g0[3] = (unsigned)((ga >> 32) & 0x01FFFFFFu) | (2u << 30); // addr[56:32] | type=2
    unsigned td0 = (unsigned)K;                        // tensor_dim0 (elements)
    unsigned td1 = 16u;                                // tensor_dim1 (rows)
    unsigned long long st0 = (unsigned long long)lda;  // dim0 stride (elements)
    i32x8 g1;
    g1[0] = (int)(1u << 16);                           // data_size=1 (2 bytes)
    g1[1] = (int)((td0 & 0xFFFFu) << 16);              // tensor_dim0[15:0]
    g1[2] = (int)((td0 >> 16) | ((td1 & 0xFFFFu) << 16));
    g1[3] = (int)((td1 >> 16) | ((unsigned)K << 16));  // tile_dim0 = K
    g1[4] = (int)16;                                   // tile_dim1 = 16, tile_dim2 = 0
    g1[5] = (int)(st0 & 0xFFFFFFFFu);
    g1[6] = (int)((st0 >> 32) & 0xFFFFu);
    g1[7] = 0;
    i32x4 gz = {0, 0, 0, 0};
#if defined(__clang_major__) && (__clang_major__ >= 23)
    i32x8 gz8 = {0, 0, 0, 0, 0, 0, 0, 0};
    __builtin_amdgcn_tensor_load_to_lds(g0, g1, gz, gz, gz8, 0);
#else
    __builtin_amdgcn_tensor_load_to_lds(g0, g1, gz, gz, 0);
#endif
    __builtin_amdgcn_s_wait_tensorcnt(0);
  }
  __syncthreads();
#else
  // fallback: vectorized cooperative copy (all K are multiples of 16)
  for (int e = tid * 8; e < 16 * K; e += blockDim.x * 8) {
    int row = e / K, col = e % K;
    *(u32x4*)&Aslab[e] = *(const u32x4*)&A[(long)(mt * 16 + row) * lda + col];
  }
  __syncthreads();
#endif

  if (nt >= ntiles) return; // no barriers below: out-of-range waves exit now

  const bf16* arow = &Aslab[(long)m * K];
  v8f acc = {};
  for (int kc = 0; kc < K; kc += 32) {
    FragU ua;
    ua.u[0] = *(const u32x4*)(arow + kc + kq);
    ua.u[1] = *(const u32x4*)(arow + kc + 16 + kq);
    v16bf wf = load_frag_g(W + kc, nt * 16, ldw, lane);
    __builtin_prefetch(&W[(long)(nt * 16 + (lane & 15)) * ldw + kc + 32], 0, 0);
    acc = wmma_bf16(ua.v, wf, acc);
  }
  int n = nt * 16 + (lane & 15);
  int rb = (lane >> 4) << 3;
  float bv = bias ? bias[n] : 0.f;
  int act = (flags >> 2) & 7;
#pragma unroll
  for (int r = 0; r < 8; ++r) {
    int mm = mt * 16 + rb + r;
    float v = acc[r] + bv;
    if (flags & 1) v += C[(long)mm * ldc + n];
    if (act == 1) v = fmaxf(v, 0.f);
    else if (act == 2) v = (v > 0.f) ? v : 0.1f * v;
    else if (act == 3) v = sigf(v);
    if (C)  C[(long)mm * ldc + n] = v;
    if (Cb) Cb[(long)mm * ldcb + n] = f2bf(v);
  }
}

// ---------------------------------------------------------------------------
// Implicit-GEMM 1D conv, k=5, pad=2, 512->512 channels, relu, optional posenc.
// X,Y: [B,T,512] bf16.  Wc: [k][oc][ic] bf16 (layer already selected).
// grid = (8, B*T/16), block = 128 (4 waves -> 4 oc-tiles).
// ---------------------------------------------------------------------------
__global__ void __launch_bounds__(128)
conv1d_wmma(const bf16* __restrict__ X, const bf16* __restrict__ Wc,
            const float* __restrict__ bias, bf16* __restrict__ Y, int addpos) {
  int lane = threadIdx.x & 31, wave = threadIdx.x >> 5;
  int mt = blockIdx.y;
  int nt = blockIdx.x * 4 + wave;
  int m0 = mt * 16;
  int b = m0 >> 10, t0 = m0 & (TLEN - 1);
  int m = lane & 15, kq = (lane >> 4) << 3;
  v8f acc = {};
  for (int k = 0; k < 5; ++k) {
    int t = t0 + m + k - 2;
    bool ok = (t >= 0) && (t < TLEN);
    for (int icc = 0; icc < DM / 32; ++icc) {
      v16bf af;
      if (ok) {
        const bf16* r = X + ((long)(b * TLEN + t) * DM + icc * 32);
        FragU ua;
        ua.u[0] = *(const u32x4*)(r + kq);
        ua.u[1] = *(const u32x4*)(r + 16 + kq);
        af = ua.v;
      } else {
        af = frag_zero();
      }
      v16bf wf = load_frag_g(Wc + (long)k * DM * DM + icc * 32, nt * 16, DM, lane);
      acc = wmma_bf16(af, wf, acc);
    }
  }
  int n = nt * 16 + (lane & 15);
  int rb = (lane >> 4) << 3;
  float bv = bias[n];
#pragma unroll
  for (int r = 0; r < 8; ++r) {
    int mm = m0 + rb + r;
    int t = mm & (TLEN - 1);
    float v = fmaxf(acc[r] + bv, 0.f);
    if (addpos) {
      float div = __expf((float)((n >> 1) * 2) * (-9.210340371976184f / (float)DM));
      float ang = (float)t * div;
      v += (n & 1) ? __cosf(ang) : __sinf(ang);
    }
    Y[(long)mm * DM + n] = f2bf(v);
  }
}

// ---------------------------------------------------------------------------
// Fused location-attention energy + softmax.  One block per (head,batch),
// 512 threads (16 waves).  fl = F(t,c) @ locl_w(a,c)^T is one single-K-chunk
// WMMA per 16-t tile (CC=32) and never hits memory; then
// e[t] = v . tanh(q + keys + fl) / sqrt(DA), softmax over T in LDS.
// keysT is pre-transposed to [B][NH][DA][T] so each (a, t-tile) read is one
// 16-byte load of 8 consecutive t.
// ---------------------------------------------------------------------------
__global__ void __launch_bounds__(512)
attn_energy(const bf16* __restrict__ fbf,   // [NH][B][T][32]
            const bf16* __restrict__ wlocl, // [NH][512][32]
            const float* __restrict__ loclb,// [NH][512]
            const float* __restrict__ qv,   // [B][NH*512]
            const bf16* __restrict__ keysT, // [B][NH][DA][T]
            const float* __restrict__ vw,   // [NH][512]
            const float* __restrict__ vb,   // [NH]
            float* __restrict__ alpha) {    // [NH][B][T]
  int hb = blockIdx.x;
  int h = hb >> 5, b = hb & 31;
  int tid = threadIdx.x;
  int lane = tid & 31, wave = tid >> 5;
  __shared__ float es[TLEN];
  __shared__ float sred[32];
  const float rscale = 0.044194173824159216f; // 1/sqrt(512)
  int m = lane & 15, kq = (lane >> 4) << 3, half = lane >> 4;
  for (int tt = wave; tt < TLEN / 16; tt += 16) {
    int t0 = tt * 16;
    const bf16* fr = fbf + ((long)hb * TLEN + t0 + m) * CCH;
    FragU ua;
    ua.u[0] = *(const u32x4*)(fr + kq);
    ua.u[1] = *(const u32x4*)(fr + 16 + kq);
    v16bf af = ua.v;
    float part[8];
#pragma unroll
    for (int r = 0; r < 8; ++r) part[r] = 0.f;
#pragma unroll 4
    for (int at = 0; at < DATT / 16; ++at) {
      v16bf wf = load_frag_g(wlocl + (long)h * DATT * CCH, at * 16, CCH, lane);
      v8f c = {};
      c = wmma_bf16(af, wf, c);
      int a = at * 16 + (lane & 15);
      float qa = qv[b * (NHEAD * DATT) + h * DATT + a] + loclb[h * DATT + a];
      float va = vw[h * DATT + a];
      union { u32x4 u; bf16 e[8]; } ku;
      ku.u = *(const u32x4*)(keysT + (((long)(b * NHEAD + h) * DATT + a) * TLEN + t0 + half * 8));
#pragma unroll
      for (int r = 0; r < 8; ++r) {
        float s = c[r] + qa + bf2f(ku.e[r]);
        part[r] += tanhf(s) * va;
      }
    }
#pragma unroll
    for (int r = 0; r < 8; ++r) {
      float x = part[r];
      x += __shfl_xor(x, 8, 16);
      x += __shfl_xor(x, 4, 16);
      x += __shfl_xor(x, 2, 16);
      x += __shfl_xor(x, 1, 16);
      part[r] = x;
    }
    if ((lane & 15) == 0) {
#pragma unroll
      for (int r = 0; r < 8; ++r) es[t0 + half * 8 + r] = (part[r] + vb[h]) * rscale;
    }
  }
  __syncthreads();
  // softmax over 1024
  float e0 = es[tid], e1 = es[tid + 512];
  float lm = fmaxf(e0, e1);
#pragma unroll
  for (int o = 16; o >= 1; o >>= 1) lm = fmaxf(lm, __shfl_xor(lm, o, 32));
  if (lane == 0) sred[wave] = lm;
  __syncthreads();
  if (tid < 32) {
    float x = (tid < 16) ? sred[tid] : -3.0e38f;
#pragma unroll
    for (int o = 8; o >= 1; o >>= 1) x = fmaxf(x, __shfl_xor(x, o, 32));
    if (tid == 0) sred[16] = x;
  }
  __syncthreads();
  float mx = sred[16];
  float x0 = __expf(e0 - mx), x1 = __expf(e1 - mx);
  float ls = x0 + x1;
#pragma unroll
  for (int o = 16; o >= 1; o >>= 1) ls += __shfl_xor(ls, o, 32);
  if (lane == 0) sred[wave] = ls;
  __syncthreads();
  if (tid < 32) {
    float x = (tid < 16) ? sred[tid] : 0.f;
#pragma unroll
    for (int o = 8; o >= 1; o >>= 1) x += __shfl_xor(x, o, 32);
    if (tid == 0) sred[17] = x;
  }
  __syncthreads();
  float inv = 1.f / sred[17];
  float* ao = alpha + (long)hb * TLEN;
  ao[tid] = x0 * inv;
  ao[tid + 512] = x1 * inv;
}

// keys [B*T][NH*DA] -> keysT [B][NH][DA][T]
__global__ void transpose_keys(const bf16* __restrict__ in, bf16* __restrict__ out) {
  long i = blockIdx.x * (long)blockDim.x + threadIdx.x;
  long n = (long)BSZ * TLEN * NHEAD * DATT;
  if (i >= n) return;
  int t = (int)(i & (TLEN - 1));
  long r = i >> 10;
  int a = (int)(r & (DATT - 1)); r >>= 9;
  int h = (int)(r & (NHEAD - 1));
  int b = (int)(r >> 1);
  out[i] = in[((long)(b * TLEN + t)) * (NHEAD * DATT) + h * DATT + a];
}

// location conv: f[h][b][t][c] = bias + sum_tap prev*w0 + cum*w1
__global__ void loc_conv(const float* __restrict__ prev_a, const float* __restrict__ cum_a,
                         const float* __restrict__ w, const float* __restrict__ bias,
                         bf16* __restrict__ f) {
  long i = blockIdx.x * (long)blockDim.x + threadIdx.x;
  if (i >= (long)NHEAD * BSZ * TLEN * CCH) return;
  int cc = (int)(i & 31);
  long r = i >> 5;
  int t = (int)(r & (TLEN - 1)); r >>= 10;
  int b = (int)(r & 31);
  int h = (int)(r >> 5);
  int oc = h * CCH + cc;
  const float* wv = w + (long)oc * 2 * CKER;
  float acc = bias[oc];
  const float* pa = prev_a + (long)b * TLEN;
  const float* ca = cum_a + (long)b * TLEN;
  for (int tap = 0; tap < CKER; ++tap) {
    int tt = t + tap - CKER / 2;
    if (tt >= 0 && tt < TLEN) acc += pa[tt] * wv[tap] + ca[tt] * wv[CKER + tap];
  }
  f[i] = f2bf(acc);
}

__global__ void attn_update(const float* __restrict__ alpha, const float* __restrict__ oaw,
                            const float* __restrict__ oab, float* __restrict__ prev_a,
                            float* __restrict__ cum_a) {
  int i = blockIdx.x * blockDim.x + threadIdx.x;
  if (i >= BSZ * TLEN) return;
  float aw = alpha[i] * oaw[0] + alpha[i + BSZ * TLEN] * oaw[1] + oab[0];
  prev_a[i] = aw;
  cum_a[i] += aw;
}

// ctxs[h][b][d] = sum_t alpha * memory, written straight into cat layout [b][h*D+d]
__global__ void ctx_kernel(const float* __restrict__ alpha, const bf16* __restrict__ mem,
                           bf16* __restrict__ ctxcat) {
  int hb = blockIdx.x;
  int h = hb >> 5, b = hb & 31;
  int d = threadIdx.x; // 512
  const float* al = alpha + (long)hb * TLEN;
  const bf16* mb = mem + (long)b * TLEN * DM;
  float acc = 0.f;
  for (int t = 0; t < TLEN; ++t) acc += al[t] * bf2f(mb[(long)t * DM + d]);
  ctxcat[b * (NHEAD * DM) + h * DM + d] = f2bf(acc);
}

__global__ void metagate_ln(const float* __restrict__ q, const float* __restrict__ ctx,
                            const float* __restrict__ gamma, const float* __restrict__ beta,
                            const float* __restrict__ lng, const float* __restrict__ lnb,
                            bf16* __restrict__ out) {
  int b = blockIdx.x, tid = threadIdx.x; // 512 threads
  __shared__ float sred[32];
  int j0 = tid, j1 = tid + 512;
  float g0 = q[b * 512 + j0];
  float g1 = ctx[b * 512 + (j1 - 512)];
  g0 = g0 * (1.f + gamma[b * 1024 + j0]) + beta[b * 1024 + j0];
  g1 = g1 * (1.f + gamma[b * 1024 + j1]) + beta[b * 1024 + j1];
  float s = g0 + g1, s2 = g0 * g0 + g1 * g1;
  int lane = tid & 31, wv = tid >> 5;
#pragma unroll
  for (int o = 16; o >= 1; o >>= 1) { s += __shfl_xor(s, o, 32); s2 += __shfl_xor(s2, o, 32); }
  if (lane == 0) { sred[wv] = s; sred[16 + wv] = s2; }
  __syncthreads();
  if (tid == 0) {
    float a = 0.f, c = 0.f;
    for (int i = 0; i < 16; ++i) { a += sred[i]; c += sred[16 + i]; }
    sred[0] = a; sred[1] = c;
  }
  __syncthreads();
  float mu = sred[0] / 1024.f;
  float var = sred[1] / 1024.f - mu * mu;
  float inv = rsqrtf(var + 1e-5f);
  out[b * 1024 + j0] = f2bf((g0 - mu) * inv * lng[j0] + lnb[j0]);
  out[b * 1024 + j1] = f2bf((g1 - mu) * inv * lng[j1] + lnb[j1]);
}

__global__ void lstm_pointwise(const float* __restrict__ gates, const float* __restrict__ bih,
                               const float* __restrict__ bhh, float* __restrict__ c,
                               float* __restrict__ hf, bf16* __restrict__ hb, int rows) {
  long i = blockIdx.x * (long)blockDim.x + threadIdx.x;
  if (i >= (long)rows * DM) return;
  int d = (int)(i % DM);
  long r = i / DM;
  const float* g = gates + r * 4 * DM;
  float gi = sigf(g[d] + bih[d] + bhh[d]);
  float gf = sigf(g[DM + d] + bih[DM + d] + bhh[DM + d]);
  float gg = tanhf(g[2 * DM + d] + bih[2 * DM + d] + bhh[2 * DM + d]);
  float go = sigf(g[3 * DM + d] + bih[3 * DM + d] + bhh[3 * DM + d]);
  float c2 = gf * c[i] + gi * gg;
  float h2 = go * tanhf(c2);
  c[i] = c2;
  if (hf) hf[i] = h2;
  hb[i] = f2bf(h2);
}

__global__ void embed_tokens(const float* __restrict__ truth,
                             const float* __restrict__ xw, const float* __restrict__ xb,
                             const float* __restrict__ yw, const float* __restrict__ yb,
                             const float* __restrict__ dw, const float* __restrict__ db,
                             const float* __restrict__ etype,
                             bf16* __restrict__ tok) { // [4][B*S][512]
  long i = blockIdx.x * (long)blockDim.x + threadIdx.x;
  if (i >= (long)BSZ * SLEN * DM) return;
  int d = (int)(i % DM);
  long bs = i / DM;
  int s = (int)(bs % SLEN);
  int b = (int)(bs / SLEN);
  float px = 0.f, py = 0.f, pdt = 0.f;
  int ti = 0;
  if (s > 0) {
    const float* pv = truth + ((long)b * SLEN + (s - 1)) * 4;
    px = pv[0]; py = pv[1]; pdt = pv[2]; ti = (int)pv[3];
  }
  long o = bs * DM + d;
  long NS = (long)BSZ * SLEN * DM;
  tok[o]          = f2bf(px * xw[d] + xb[d]);
  tok[o + NS]     = f2bf(py * yw[d] + yb[d]);
  tok[o + 2 * NS] = f2bf(etype[ti * DM + d]);
  tok[o + 3 * NS] = f2bf(pdt * dw[d] + db[d]);
}

__global__ void meta_gate_params(const float* __restrict__ meta, const float* __restrict__ gw,
                                 const float* __restrict__ gb, const float* __restrict__ bw,
                                 const float* __restrict__ bb, float* __restrict__ gamma,
                                 float* __restrict__ beta) {
  int i = blockIdx.x * blockDim.x + threadIdx.x;
  if (i >= BSZ * 2 * DM) return;
  int j = i % (2 * DM);
  int b = i / (2 * DM);
  const float* mv = meta + b * 64;
  float s1 = gb[j], s2 = bb[j];
  for (int k = 0; k < 64; ++k) { s1 += mv[k] * gw[j * 64 + k]; s2 += mv[k] * bw[j * 64 + k]; }
  gamma[i] = tanhf(s1);
  beta[i] = tanhf(s2);
}

__global__ void head_out(const bf16* __restrict__ h2, const float* __restrict__ w3,
                         const float* __restrict__ b3, float* __restrict__ out,
                         int odim, int step, int act) {
  int i = blockIdx.x * blockDim.x + threadIdx.x;
  if (i >= BSZ * odim) return;
  int o = i % odim, b = i / odim;
  float s = b3[o];
  const bf16* hr = h2 + b * DM;
  const float* wr = w3 + o * DM;
  for (int k = 0; k < DM; ++k) s += bf2f(hr[k]) * wr[k];
  if (act == 3) s = sigf(s);
  out[((long)b * SLEN + step) * odim + o] = s;
}

__global__ void cvt_f32_to_bf16(const float* __restrict__ in, bf16* __restrict__ out, long n) {
  long i = blockIdx.x * (long)blockDim.x + threadIdx.x;
  if (i < n) out[i] = f2bf(in[i]);
}

// row-major [rows][ic] f32 -> [rows][oc] bf16, zero-padded cols ic..oc
__global__ void cvt_pad(const float* __restrict__ in, bf16* __restrict__ out,
                        long rows, int ic, int oc) {
  long i = blockIdx.x * (long)blockDim.x + threadIdx.x;
  if (i >= rows * oc) return;
  int c = (int)(i % oc);
  long r = i / oc;
  out[i] = (c < ic) ? f2bf(in[r * ic + c]) : f2bf(0.f);
}

// [5][oc][ic][k] f32 -> [(l*5+k)][oc][ic] bf16
__global__ void cvt_conv_w(const float* __restrict__ in, bf16* __restrict__ out) {
  long n = (long)5 * 512 * 512 * 5;
  long i = blockIdx.x * (long)blockDim.x + threadIdx.x;
  if (i >= n) return;
  int k = (int)(i % 5);
  long r = i / 5;
  int ic = (int)(r % 512); r /= 512;
  int oc = (int)(r % 512);
  int l = (int)(r / 512);
  out[(((long)(l * 5 + k) * 512 + oc) * 512 + ic)] = f2bf(in[i]);
}

// ---------------------------------------------------------------------------
extern "C" void kernel_launch(void* const* d_in, const int* in_sizes, int n_in,
                              void* d_out, int out_size, void* d_ws, size_t ws_size,
                              hipStream_t stream) {
  (void)in_sizes; (void)n_in; (void)out_size; (void)ws_size;
#define PF(i) ((const float*)d_in[(i)])
  char* wsp = (char*)d_ws;
  size_t cur = 0;
  auto alloc = [&](size_t bytes) -> void* {
    cur = (cur + 255) & ~(size_t)255;
    void* p = wsp + cur;
    cur += bytes;
    return p;
  };
  // bf16 weight copies
  bf16* w_encproj = (bf16*)alloc((size_t)512 * KPAD * 2);
  bf16* w_conv    = (bf16*)alloc((size_t)5 * 5 * 512 * 512 * 2);
  bf16* w_pre1    = (bf16*)alloc((size_t)256 * 512 * 2);
  bf16* w_pre2    = (bf16*)alloc((size_t)256 * 256 * 2);
  bf16* w_embp_ih = (bf16*)alloc((size_t)2048 * 512 * 2);
  bf16* w_embp_hh = (bf16*)alloc((size_t)2048 * 512 * 2);
  bf16* w_q       = (bf16*)alloc((size_t)1024 * 512 * 2);
  bf16* w_k       = (bf16*)alloc((size_t)1024 * 512 * 2);
  bf16* w_locl    = (bf16*)alloc((size_t)2 * 512 * 32 * 2);
  bf16* w_attout  = (bf16*)alloc((size_t)512 * 1024 * 2);
  bf16* w_arnn_ih[4]; bf16* w_arnn_hh[4]; bf16* w_drnn_ih[4]; bf16* w_drnn_hh[4];
  int arnnK[4] = {768, 512, 512, 512};
  int drnnK[4] = {1024, 512, 512, 512};
  for (int l = 0; l < 4; ++l) {
    w_arnn_ih[l] = (bf16*)alloc((size_t)2048 * arnnK[l] * 2);
    w_arnn_hh[l] = (bf16*)alloc((size_t)2048 * 512 * 2);
    w_drnn_ih[l] = (bf16*)alloc((size_t)2048 * drnnK[l] * 2);
    w_drnn_hh[l] = (bf16*)alloc((size_t)2048 * 512 * 2);
  }
  bf16* w_h1[5]; bf16* w_h2[5];
  for (int hh = 0; hh < 5; ++hh) {
    w_h1[hh] = (bf16*)alloc((size_t)512 * 512 * 2);
    w_h2[hh] = (bf16*)alloc((size_t)512 * 512 * 2);
  }
  // activations
  bf16* a_in   = (bf16*)alloc((size_t)32768 * KPAD * 2);
  bf16* xb0    = (bf16*)alloc((size_t)32768 * 512 * 2);
  bf16* xb1    = (bf16*)alloc((size_t)32768 * 512 * 2);
  bf16* mem    = (bf16*)alloc((size_t)32768 * 512 * 2);
  bf16* keys   = (bf16*)alloc((size_t)32768 * 1024 * 2);
  bf16* keysT  = (bf16*)alloc((size_t)32768 * 1024 * 2);
  bf16* tok    = (bf16*)alloc((size_t)4 * 4096 * 512 * 2);
  bf16* embp_h = (bf16*)alloc((size_t)4096 * 512 * 2);
  float* embp_c = (float*)alloc((size_t)4096 * 512 * 4);
  float* gates  = (float*)alloc((size_t)4096 * 2048 * 4);
  float* qbuf   = (float*)alloc((size_t)32 * 1024 * 4);
  bf16* pre1_bf = (bf16*)alloc((size_t)32 * 256 * 2);
  bf16* inbuf   = (bf16*)alloc((size_t)32 * 768 * 2);
  bf16* ah_bf   = (bf16*)alloc((size_t)4 * 32 * 512 * 2);
  float* ah_f   = (float*)alloc((size_t)32 * 512 * 4); // layer3 query (f32)
  float* ac     = (float*)alloc((size_t)4 * 32 * 512 * 4);
  bf16* dh_bf   = (bf16*)alloc((size_t)4 * 32 * 512 * 2);
  float* dc     = (float*)alloc((size_t)4 * 32 * 512 * 4);
  float* ctx_f  = (float*)alloc((size_t)32 * 512 * 4);
  float* prev_a = (float*)alloc((size_t)32 * 1024 * 4);
  float* cum_a  = (float*)alloc((size_t)32 * 1024 * 4);
  bf16* fbf     = (bf16*)alloc((size_t)2 * 32 * 1024 * 32 * 2);
  float* alpha  = (float*)alloc((size_t)2 * 32 * 1024 * 4);
  bf16* ctxcat  = (bf16*)alloc((size_t)32 * 1024 * 2);
  bf16* gbuf    = (bf16*)alloc((size_t)32 * 1024 * 2);
  float* gamma  = (float*)alloc((size_t)32 * 1024 * 4);
  float* beta   = (float*)alloc((size_t)32 * 1024 * 4);
  bf16* h1      = (bf16*)alloc((size_t)32 * 512 * 2);
  bf16* h2      = (bf16*)alloc((size_t)32 * 512 * 2);

  auto cvt = [&](const float* src, bf16* dst, long n) {
    cvt_f32_to_bf16<<<dim3((unsigned)((n + 255) / 256)), 256, 0, stream>>>(src, dst, n);
  };
  // padded conversions for the K=80 GEMM
  cvt_pad<<<dim3((unsigned)((512L * KPAD + 255) / 256)), 256, 0, stream>>>(
      PF(4), w_encproj, 512, INDIM, KPAD);
  cvt_pad<<<dim3((unsigned)((32768L * KPAD + 255) / 256)), 256, 0, stream>>>(
      PF(0), a_in, 32768, INDIM, KPAD);
  cvt_conv_w<<<dim3((unsigned)((6553600L + 255) / 256)), 256, 0, stream>>>(PF(6), w_conv);
  cvt(PF(8), w_pre1, 256L * 512);
  cvt(PF(10), w_pre2, 256L * 256);
  cvt(PF(25), w_embp_ih, 2048L * 512);
  cvt(PF(26), w_embp_hh, 2048L * 512);
  cvt(PF(59), w_q, 1024L * 512);
  cvt(PF(61), w_k, 1024L * 512);
  cvt(PF(65), w_locl, 2L * 512 * 32);
  cvt(PF(69), w_attout, 512L * 1024);
  for (int l = 0; l < 4; ++l) {
    cvt(PF(73 + l), w_arnn_ih[l], 2048L * arnnK[l]);
    cvt(PF(77 + l), w_arnn_hh[l], 2048L * 512);
    cvt(PF(89 + l), w_drnn_ih[l], 2048L * drnnK[l]);
    cvt(PF(93 + l), w_drnn_hh[l], 2048L * 512);
  }
  for (int hh = 0; hh < 5; ++hh) {
    int base = 29 + hh * 6;
    cvt(PF(base + 0), w_h1[hh], 512L * 512);
    cvt(PF(base + 2), w_h2[hh], 512L * 512);
  }

  // state init
  hipMemsetAsync(inbuf, 0, (size_t)32 * 768 * 2, stream);
  hipMemsetAsync(ah_bf, 0, (size_t)4 * 32 * 512 * 2, stream);
  hipMemsetAsync(ah_f, 0, (size_t)32 * 512 * 4, stream);
  hipMemsetAsync(ac, 0, (size_t)4 * 32 * 512 * 4, stream);
  hipMemsetAsync(dh_bf, 0, (size_t)4 * 32 * 512 * 2, stream);
  hipMemsetAsync(dc, 0, (size_t)4 * 32 * 512 * 4, stream);
  hipMemsetAsync(prev_a, 0, (size_t)32 * 1024 * 4, stream);
  hipMemsetAsync(cum_a, 0, (size_t)32 * 1024 * 4, stream);
  hipMemsetAsync(embp_h, 0, (size_t)4096 * 512 * 2, stream);
  hipMemsetAsync(embp_c, 0, (size_t)4096 * 512 * 4, stream);

  auto gemm = [&](const bf16* A, int lda, const bf16* W, int ldw, const float* bias,
                  float* C, int ldc, bf16* Cb, int ldcb, int M, int N, int K,
                  int accf, int act) {
    dim3 g((unsigned)((N / 16 + 3) / 4), (unsigned)(M / 16));
    gemm_bf16_wmma<<<g, 128, 0, stream>>>(A, lda, W, ldw, bias, C, ldc, Cb, ldcb,
                                          M, N, K, accf | (act << 2));
  };

  // ---- encoder ----
  gemm(a_in, KPAD, w_encproj, KPAD, PF(5), nullptr, 0, xb0, 512, 32768, 512, KPAD, 0, 0);
  bf16* cin = xb0; bf16* cout = xb1;
  for (int l = 0; l < 5; ++l) {
    bf16* dst = (l == 4) ? mem : cout;
    conv1d_wmma<<<dim3(8, 2048), 128, 0, stream>>>(cin, w_conv + (size_t)l * 5 * 512 * 512,
                                                   PF(7) + l * 512, dst, (l == 4) ? 1 : 0);
    bf16* t = cin; cin = cout; cout = t;
  }
  // keys for both heads in one GEMM, then transpose for the energy kernel
  gemm(mem, 512, w_k, 512, PF(62), nullptr, 0, keys, 1024, 32768, 1024, 512, 0, 0);
  transpose_keys<<<dim3((unsigned)(((long)32768 * 1024 + 255) / 256)), 256, 0, stream>>>(keys, keysT);

  // ---- decoder input embedding (4-token LSTM) ----
  embed_tokens<<<dim3((unsigned)((2097152L + 255) / 256)), 256, 0, stream>>>(
      PF(1), PF(18), PF(19), PF(20), PF(21), PF(22), PF(23), PF(24), tok);
  for (int k = 0; k < 4; ++k) {
    gemm(tok + (size_t)k * 4096 * 512, 512, w_embp_ih, 512, nullptr, gates, 2048,
         nullptr, 0, 4096, 2048, 512, 0, 0);
    gemm(embp_h, 512, w_embp_hh, 512, nullptr, gates, 2048, nullptr, 0, 4096, 2048, 512, 1, 0);
    lstm_pointwise<<<dim3((unsigned)((4096L * 512 + 255) / 256)), 256, 0, stream>>>(
        gates, PF(27), PF(28), embp_c, nullptr, embp_h, 4096);
  }
  // gamma/beta (step-invariant)
  meta_gate_params<<<dim3((32768 + 255) / 256), 256, 0, stream>>>(
      PF(3), PF(12), PF(13), PF(14), PF(15), gamma, beta);

  struct HeadDesc { int base; long ooff; int odim; int act; };
  HeadDesc H[5] = { {29, 45056, 1, 3}, {35, 0, 1, 3}, {41, 4096, 1, 3},
                    {47, 8192, 1, 0}, {53, 12288, 8, 0} };

  // ---- sequential decoder scan ----
  for (int s = 0; s < SLEN; ++s) {
    // prenet (leaky 0.1); pre goes into inbuf[:,512:768]
    gemm(embp_h + (size_t)s * 512, SLEN * 512, w_pre1, 512, PF(9), nullptr, 0,
         pre1_bf, 256, 32, 256, 512, 0, 2);
    gemm(pre1_bf, 256, w_pre2, 256, PF(11), nullptr, 0, inbuf + 512, 768,
         32, 256, 256, 0, 2);
    // attention LSTM stack
    const bf16* x = inbuf; int xld = 768, xK = 768;
    for (int l = 0; l < 4; ++l) {
      gemm(x, xld, w_arnn_ih[l], xK, nullptr, gates, 2048, nullptr, 0, 32, 2048, xK, 0, 0);
      gemm(ah_bf + (size_t)l * 32 * 512, 512, w_arnn_hh[l], 512, nullptr, gates, 2048,
           nullptr, 0, 32, 2048, 512, 1, 0);
      lstm_pointwise<<<dim3((32 * 512 + 255) / 256), 256, 0, stream>>>(
          gates, PF(81 + l), PF(85 + l), ac + (size_t)l * 32 * 512,
          (l == 3) ? ah_f : nullptr, ah_bf + (size_t)l * 32 * 512, 32);
      x = ah_bf + (size_t)l * 32 * 512; xld = 512; xK = 512;
    }
    // q projection (both heads), location conv, fused energy+softmax
    gemm(ah_bf + (size_t)3 * 32 * 512, 512, w_q, 512, PF(60), qbuf, 1024,
         nullptr, 0, 32, 1024, 512, 0, 0);
    loc_conv<<<dim3((unsigned)((2097152L + 255) / 256)), 256, 0, stream>>>(
        prev_a, cum_a, PF(63), PF(64), fbf);
    attn_energy<<<64, 512, 0, stream>>>(fbf, w_locl, PF(66), qbuf, keysT, PF(67), PF(68), alpha);
    attn_update<<<dim3(32768 / 256), 256, 0, stream>>>(alpha, PF(71), PF(72), prev_a, cum_a);
    ctx_kernel<<<64, 512, 0, stream>>>(alpha, mem, ctxcat);
    // att_out -> ctx (f32 for metagate) and bf16 into inbuf[:,0:512] for next step
    gemm(ctxcat, 1024, w_attout, 1024, PF(70), ctx_f, 512, inbuf, 768, 32, 512, 1024, 0, 0);
    metagate_ln<<<32, 512, 0, stream>>>(ah_f, ctx_f, gamma, beta, PF(16), PF(17), gbuf);
    // decoder LSTM stack
    x = gbuf; xld = 1024; xK = 1024;
    for (int l = 0; l < 4; ++l) {
      gemm(x, xld, w_drnn_ih[l], xK, nullptr, gates, 2048, nullptr, 0, 32, 2048, xK, 0, 0);
      gemm(dh_bf + (size_t)l * 32 * 512, 512, w_drnn_hh[l], 512, nullptr, gates, 2048,
           nullptr, 0, 32, 2048, 512, 1, 0);
      lstm_pointwise<<<dim3((32 * 512 + 255) / 256), 256, 0, stream>>>(
          gates, PF(97 + l), PF(101 + l), dc + (size_t)l * 32 * 512,
          nullptr, dh_bf + (size_t)l * 32 * 512, 32);
      x = dh_bf + (size_t)l * 32 * 512; xld = 512; xK = 512;
    }
    // output heads
    for (int hh = 0; hh < 5; ++hh) {
      gemm(dh_bf + (size_t)3 * 32 * 512, 512, w_h1[hh], 512, PF(H[hh].base + 1),
           nullptr, 0, h1, 512, 32, 512, 512, 0, 1);
      gemm(h1, 512, w_h2[hh], 512, PF(H[hh].base + 3), nullptr, 0, h2, 512,
           32, 512, 512, 0, 1);
      head_out<<<dim3((unsigned)((32 * H[hh].odim + 63) / 64)), 64, 0, stream>>>(
          h2, PF(H[hh].base + 4), PF(H[hh].base + 5),
          (float*)d_out + H[hh].ooff, H[hh].odim, s, H[hh].act);
    }
  }
#undef PF
}
